// DentateCircuit_59966333387259
// MI455X (gfx1250) — matrix operand order repaired and checked by reading, not compiled, not run on hardware
//
#include <hip/hip_runtime.h>

// ---------------------------------------------------------------------------
// CDNA5 (gfx1250) dentate-gyrus circuit step.
//   - f32 weights converted once per call into WMMA-B-tiled f16 in d_ws
//     (~186 MB, sized to be L2-resident on MI455X's 192 MB L2).
//   - Each GEMV chunk = v_wmma_f32_16x16x32_f16 with A row0 = spike chunk,
//     B = 32x16 weight tile (1 KB, one coalesced 32B load per lane).
//   - Plain (rotation-free) inner loop: only VMEM loads rewrite the WMMA
//     operand registers (scoreboarded by loadcnt, no hazard NOPs); unroll 2
//     gives the scheduler two disjoint register sets to pipeline with.
//   - Scalar (readfirstlane) wave ids keep loop control on the SALU and
//     EXEC all-ones through every WMMA.
//   - Recurrence handled by 3 kernels per timestep on the stream.
// ---------------------------------------------------------------------------

typedef __attribute__((ext_vector_type(16))) _Float16 v16h;
typedef __attribute__((ext_vector_type(8)))  _Float16 v8h;
typedef __attribute__((ext_vector_type(8)))  float    v8f;

#define N_GC   20000
#define N_MC   600
#define N_PV   600
#define N_SST  400
#define N_MEC  1200
#define T_STEPS 100

// K tiles (pre padded to mult of 32), N tiles (post padded to mult of 16)
#define KT_GC  625   // 20000/32
#define KT_MC  19    // 608/32
#define KT_PV  19
#define KT_SST 13    // 416/32
#define KT_MEC 38    // 1216/32
#define NT_GC  1250
#define NT_MC  38    // 608/16
#define NT_PV  38
#define NT_SST 25    // 400/16

#define V_TH_    (-50.0f)
#define V_REST_  (-70.0f)
#define V_RESET_ (-70.0f)
#define E_EXC_   (0.0f)
#define E_INH_   (-70.0f)
#define DT_      (0.1f)
#define TAU_M_   (20.0f)
#define D_AMPA_  (0.95122942450071400f)  // exp(-0.1/2)
#define D_GABA_  (0.99004983374916810f)  // exp(-0.1/10)

struct Params {
    // tiled f16 weights [nt][kt][lane(32)][16 halves]
    const _Float16 *gc_mc, *gc_pv, *gc_sst;
    const _Float16 *mc_gc, *mc_pv, *mc_sst, *mc_mc;
    const _Float16 *mec_gc, *mec_pv;
    const _Float16 *pv_gc, *pv_mc, *pv_pv, *pv_sst;
    const _Float16 *sst_gc, *sst_mc, *sst_pv, *sst_sst;
    // spike vectors, f16 (zero-padded to K tile size) + f32
    _Float16 *sg_h, *sm_h, *sp_h, *ss_h, *smec_h;
    float *sg_f, *sm_f, *sp_f, *ss_f;
    // state
    float *Vg, *Vm, *Vp, *Vs;
    float *gEg, *gEm, *gEp, *gEs;
    float *gIg, *gIm, *gIp, *gIs;
};

// ---------------------------------------------------------------------------
// GEMV tile accumulator: c += (A row0 = s[kt*32..kt*32+31]) x (B = W tile).
// A layout (16-bit 16x32): lane0 carries K0-7,K16-23; lane16 carries
// K8-15,K24-31; all lanes load the same chunk (rows 1..15 become replicas of
// row 0 -> only D row 0 is consumed, replicas are harmless and bounded).
// kt0/ktStride/KT/nt must be wave-uniform (scalar) values.
// ---------------------------------------------------------------------------
__device__ __forceinline__ v8f wmma_gemv(v8f c,
                                         const _Float16* __restrict__ Wt,
                                         const _Float16* __restrict__ sh,
                                         int KT, int nt, int lane,
                                         int kt0, int ktStride)
{
    const v16h* __restrict__ tbase = ((const v16h*)Wt) + (size_t)nt * KT * 32;
    const int aoff = (lane & 16) ? 8 : 0;
#pragma unroll 2
    for (int kt = kt0; kt < KT; kt += ktStride) {
        v16h b = tbase[(size_t)kt * 32 + lane];              // 32B coalesced
        __builtin_prefetch((const void*)(tbase + (size_t)(kt + 2 * ktStride) * 32), 0, 1);
        const _Float16* sa = sh + kt * 32 + aoff;
        v8h lo = *(const v8h*)sa;          // K {0..7} or {8..15}
        v8h hi = *(const v8h*)(sa + 16);   // K {16..23} or {24..31}
        v16h a;
#pragma unroll
        for (int i = 0; i < 8; ++i) { a[i] = lo[i]; a[i + 8] = hi[i]; }
        c = __builtin_amdgcn_wmma_f32_16x16x32_f16(false, a, false, b,
                                                   (short)0, c, false, false);
    }
    return c;
}

// ---------------------------------------------------------------------------
// One-time per call: f32 row-major W[npre][npost] -> tiled f16 (zero padded).
// Element (k,n): tile (kt=k/32, nt=n/16); lane = (n&15) | ((k&8)<<1); half = (k&7) | ((k&16)>>1).
// ---------------------------------------------------------------------------
__global__ void __launch_bounds__(256)
convert_kernel(const float* __restrict__ W, _Float16* __restrict__ Wt,
               int npre, int npost, int KT, int NT)
{
    size_t i = (size_t)blockIdx.x * 256 + threadIdx.x;
    size_t total = (size_t)KT * NT * 512;
    if (i >= total) return;
    size_t tile = i >> 9;
    int e    = (int)(i & 511);
    int lane = e >> 4;
    int slot = e & 15;
    int nt = (int)(tile / KT);
    int kt = (int)(tile % KT);
    int k = kt * 32 + ((slot & 8) << 1) + ((lane & 16) >> 1) + (slot & 7);
    int n = nt * 16 + (lane & 15);
    float v = (k < npre && n < npost) ? W[(size_t)k * npost + n] : 0.0f;
    Wt[i] = (_Float16)v;
}

__global__ void __launch_bounds__(256)
init_kernel(Params p)
{
    int i0 = blockIdx.x * 256 + threadIdx.x;
    int stride = gridDim.x * 256;
    for (int j = i0; j < N_GC;  j += stride) { p.Vg[j]=V_REST_; p.gEg[j]=0.f; p.gIg[j]=0.f; p.sg_f[j]=0.f; }
    for (int j = i0; j < N_MC;  j += stride) { p.Vm[j]=V_REST_; p.gEm[j]=0.f; p.gIm[j]=0.f; p.sm_f[j]=0.f; }
    for (int j = i0; j < N_PV;  j += stride) { p.Vp[j]=V_REST_; p.gEp[j]=0.f; p.gIp[j]=0.f; p.sp_f[j]=0.f; }
    for (int j = i0; j < N_SST; j += stride) { p.Vs[j]=V_REST_; p.gEs[j]=0.f; p.gIs[j]=0.f; p.ss_f[j]=0.f; }
    for (int j = i0; j < KT_GC  * 32; j += stride) p.sg_h[j]   = (_Float16)0.f;
    for (int j = i0; j < KT_MC  * 32; j += stride) p.sm_h[j]   = (_Float16)0.f;
    for (int j = i0; j < KT_PV  * 32; j += stride) p.sp_h[j]   = (_Float16)0.f;
    for (int j = i0; j < KT_SST * 32; j += stride) p.ss_h[j]   = (_Float16)0.f;
    for (int j = i0; j < KT_MEC * 32; j += stride) p.smec_h[j] = (_Float16)0.f;
}

// spikes from V at the start of the step (surrogate sigmoid), f32 + f16 copies
__global__ void __launch_bounds__(256)
spike_kernel(Params p, const float* __restrict__ mec_row)
{
    int i = blockIdx.x * 256 + threadIdx.x;
    if (i < N_GC) {
        float v = p.Vg[i];
        float s = 1.f / (1.f + __expf((V_TH_ - v) * 0.5f));
        p.sg_f[i] = s; p.sg_h[i] = (_Float16)s;
    } else if (i < N_GC + N_MC) {
        int j = i - N_GC; float v = p.Vm[j];
        float s = 1.f / (1.f + __expf((V_TH_ - v) * 0.5f));
        p.sm_f[j] = s; p.sm_h[j] = (_Float16)s;
    } else if (i < N_GC + N_MC + N_PV) {
        int j = i - N_GC - N_MC; float v = p.Vp[j];
        float s = 1.f / (1.f + __expf((V_TH_ - v) * 0.5f));
        p.sp_f[j] = s; p.sp_h[j] = (_Float16)s;
    } else if (i < N_GC + N_MC + N_PV + N_SST) {
        int j = i - N_GC - N_MC - N_PV; float v = p.Vs[j];
        float s = 1.f / (1.f + __expf((V_TH_ - v) * 0.5f));
        p.ss_f[j] = s; p.ss_h[j] = (_Float16)s;
    } else if (i < N_GC + N_MC + N_PV + N_SST + N_MEC) {
        int j = i - (N_GC + N_MC + N_PV + N_SST);
        p.smec_h[j] = (_Float16)mec_row[j];
    }
}

__device__ __forceinline__ float lif_update(float V, float ge, float gi, float s)
{
    float I  = ge * (E_EXC_ - V) + gi * (E_INH_ - V);
    float Vn = V + DT_ * ((V_REST_ - V) / TAU_M_ + I);
    return Vn - (Vn - V_RESET_) * s;
}

// MC / PV / SST: one block per 16-output tile; 8 waves split the K reduction
__global__ void __launch_bounds__(256)
inh_step_kernel(Params p)
{
    __shared__ float red[2][8][16];
    int job  = blockIdx.x;
    int wave = __builtin_amdgcn_readfirstlane(threadIdx.x >> 5);  // scalar
    int lane = threadIdx.x & 31;
    int pop, nt;
    if      (job < NT_MC)         { pop = 0; nt = job; }
    else if (job < NT_MC + NT_PV) { pop = 1; nt = job - NT_MC; }
    else                          { pop = 2; nt = job - NT_MC - NT_PV; }

    v8f ce = {}; v8f ci = {};
    if (pop == 0) {
        ce = wmma_gemv(ce, p.gc_mc,  p.sg_h,   KT_GC,  nt, lane, wave, 8);
        ce = wmma_gemv(ce, p.mc_mc,  p.sm_h,   KT_MC,  nt, lane, wave, 8);
        ci = wmma_gemv(ci, p.pv_mc,  p.sp_h,   KT_PV,  nt, lane, wave, 8);
        ci = wmma_gemv(ci, p.sst_mc, p.ss_h,   KT_SST, nt, lane, wave, 8);
    } else if (pop == 1) {
        ce = wmma_gemv(ce, p.gc_pv,  p.sg_h,   KT_GC,  nt, lane, wave, 8);
        ce = wmma_gemv(ce, p.mc_pv,  p.sm_h,   KT_MC,  nt, lane, wave, 8);
        ce = wmma_gemv(ce, p.mec_pv, p.smec_h, KT_MEC, nt, lane, wave, 8);
        ci = wmma_gemv(ci, p.pv_pv,  p.sp_h,   KT_PV,  nt, lane, wave, 8);
        ci = wmma_gemv(ci, p.sst_pv, p.ss_h,   KT_SST, nt, lane, wave, 8);
    } else {
        ce = wmma_gemv(ce, p.gc_sst,  p.sg_h,  KT_GC,  nt, lane, wave, 8);
        ce = wmma_gemv(ce, p.mc_sst,  p.sm_h,  KT_MC,  nt, lane, wave, 8);
        ci = wmma_gemv(ci, p.pv_sst,  p.sp_h,  KT_PV,  nt, lane, wave, 8);
        ci = wmma_gemv(ci, p.sst_sst, p.ss_h,  KT_SST, nt, lane, wave, 8);
    }
    if (lane < 16) { red[0][wave][lane] = ce[0]; red[1][wave][lane] = ci[0]; }
    __syncthreads();
    if (threadIdx.x < 16) {
        float se = 0.f, si = 0.f;
        for (int w = 0; w < 8; ++w) { se += red[0][w][threadIdx.x]; si += red[1][w][threadIdx.x]; }
        int j = nt * 16 + threadIdx.x;
        if (pop == 0 && j < N_MC) {
            float ge = p.gEm[j] * D_AMPA_ + se, gi = p.gIm[j] * D_GABA_ + si;
            p.gEm[j] = ge; p.gIm[j] = gi;
            p.Vm[j]  = lif_update(p.Vm[j], ge, gi, p.sm_f[j]);
        } else if (pop == 1 && j < N_PV) {
            float ge = p.gEp[j] * D_AMPA_ + se, gi = p.gIp[j] * D_GABA_ + si;
            p.gEp[j] = ge; p.gIp[j] = gi;
            p.Vp[j]  = lif_update(p.Vp[j], ge, gi, p.sp_f[j]);
        } else if (pop == 2 && j < N_SST) {
            float ge = p.gEs[j] * D_AMPA_ + se, gi = p.gIs[j] * D_GABA_ + si;
            p.gEs[j] = ge; p.gIs[j] = gi;
            p.Vs[j]  = lif_update(p.Vs[j], ge, gi, p.ss_f[j]);
        }
    }
}

// GC: one wave per 16-output tile (K is only 400..1216), fused V update + trace
__global__ void __launch_bounds__(256)
gc_step_kernel(Params p, float* __restrict__ trace_row)
{
    int wave = __builtin_amdgcn_readfirstlane(
                   blockIdx.x * 8 + (threadIdx.x >> 5));          // scalar
    int lane = threadIdx.x & 31;
    if (wave >= NT_GC) return;                 // wave-uniform guard
    v8f ce = {}; v8f ci = {};
    ce = wmma_gemv(ce, p.mc_gc,  p.sm_h,   KT_MC,  wave, lane, 0, 1);
    ce = wmma_gemv(ce, p.mec_gc, p.smec_h, KT_MEC, wave, lane, 0, 1);
    ci = wmma_gemv(ci, p.pv_gc,  p.sp_h,   KT_PV,  wave, lane, 0, 1);
    ci = wmma_gemv(ci, p.sst_gc, p.ss_h,   KT_SST, wave, lane, 0, 1);
    if (lane < 16) {
        int j = wave * 16 + lane;
        float ge = p.gEg[j] * D_AMPA_ + ce[0];
        float gi = p.gIg[j] * D_GABA_ + ci[0];
        float Vn = lif_update(p.Vg[j], ge, gi, p.sg_f[j]);
        p.gEg[j] = ge; p.gIg[j] = gi; p.Vg[j] = Vn;
        trace_row[j] = Vn;
    }
}

// ---------------------------------------------------------------------------
extern "C" void kernel_launch(void* const* d_in, const int* in_sizes, int n_in,
                              void* d_out, int out_size, void* d_ws, size_t ws_size,
                              hipStream_t stream)
{
    (void)in_sizes; (void)n_in; (void)out_size; (void)ws_size;
    // setup_inputs() dict order
    static const int pre[17]  = {20000,20000,20000,600,600,600,600,1200,1200,
                                 600,600,600,600,400,400,400,400};
    static const int post[17] = {600,600,400,20000,600,400,600,20000,600,
                                 20000,600,600,400,20000,600,600,400};

    char* ws = (char*)d_ws;
    size_t off = 0;
    auto take = [&](size_t bytes) -> char* {
        char* r = ws + off;
        off = (off + bytes + 255) & ~(size_t)255;
        return r;
    };

    _Float16* Wt[17];
    int KT[17], NT[17];
    for (int m = 0; m < 17; ++m) {
        KT[m] = (pre[m]  + 31) / 32;
        NT[m] = (post[m] + 15) / 16;
        Wt[m] = (_Float16*)take((size_t)KT[m] * NT[m] * 512 * sizeof(_Float16));
    }

    Params p;
    p.gc_mc = Wt[0];  p.gc_pv = Wt[1];  p.gc_sst = Wt[2];
    p.mc_gc = Wt[3];  p.mc_pv = Wt[4];  p.mc_sst = Wt[5]; p.mc_mc = Wt[6];
    p.mec_gc = Wt[7]; p.mec_pv = Wt[8];
    p.pv_gc = Wt[9];  p.pv_mc = Wt[10]; p.pv_pv = Wt[11]; p.pv_sst = Wt[12];
    p.sst_gc = Wt[13]; p.sst_mc = Wt[14]; p.sst_pv = Wt[15]; p.sst_sst = Wt[16];

    p.sg_h   = (_Float16*)take((size_t)KT_GC  * 32 * 2);
    p.sm_h   = (_Float16*)take((size_t)KT_MC  * 32 * 2);
    p.sp_h   = (_Float16*)take((size_t)KT_PV  * 32 * 2);
    p.ss_h   = (_Float16*)take((size_t)KT_SST * 32 * 2);
    p.smec_h = (_Float16*)take((size_t)KT_MEC * 32 * 2);
    p.sg_f = (float*)take((size_t)N_GC  * 4);
    p.sm_f = (float*)take((size_t)N_MC  * 4);
    p.sp_f = (float*)take((size_t)N_PV  * 4);
    p.ss_f = (float*)take((size_t)N_SST * 4);
    p.Vg  = (float*)take((size_t)N_GC * 4);  p.Vm  = (float*)take((size_t)N_MC * 4);
    p.Vp  = (float*)take((size_t)N_PV * 4);  p.Vs  = (float*)take((size_t)N_SST * 4);
    p.gEg = (float*)take((size_t)N_GC * 4);  p.gEm = (float*)take((size_t)N_MC * 4);
    p.gEp = (float*)take((size_t)N_PV * 4);  p.gEs = (float*)take((size_t)N_SST * 4);
    p.gIg = (float*)take((size_t)N_GC * 4);  p.gIm = (float*)take((size_t)N_MC * 4);
    p.gIp = (float*)take((size_t)N_PV * 4);  p.gIs = (float*)take((size_t)N_SST * 4);

    // one-time (per call, for determinism): f32 -> tiled f16 weights in ws
    for (int m = 0; m < 17; ++m) {
        size_t total = (size_t)KT[m] * NT[m] * 512;
        int blocks = (int)((total + 255) / 256);
        convert_kernel<<<blocks, 256, 0, stream>>>((const float*)d_in[m], Wt[m],
                                                   pre[m], post[m], KT[m], NT[m]);
    }
    init_kernel<<<96, 256, 0, stream>>>(p);

    const float* mec = (const float*)d_in[17];
    float* out = (float*)d_out;
    const int spikeBlocks = (N_GC + N_MC + N_PV + N_SST + N_MEC + 255) / 256;
    for (int t = 0; t < T_STEPS; ++t) {
        spike_kernel<<<spikeBlocks, 256, 0, stream>>>(p, mec + (size_t)t * N_MEC);
        inh_step_kernel<<<NT_MC + NT_PV + NT_SST, 256, 0, stream>>>(p);
        gc_step_kernel<<<(NT_GC + 7) / 8, 256, 0, stream>>>(p, out + (size_t)t * N_GC);
    }
}
